// FactorizedVectorQuantizer_15676630630636
// MI455X (gfx1250) — matrix operand order repaired and checked by reading, not compiled, not run on hardware
//
#include <hip/hip_runtime.h>
#include <hip/hip_bf16.h>

typedef __bf16 v16bf __attribute__((ext_vector_type(16)));
typedef __bf16 v8bf  __attribute__((ext_vector_type(8)));
typedef float  v8f   __attribute__((ext_vector_type(8)));

#define NUM_CB   4
#define CB_SIZE  1024
#define DIM      64
#define HW       4096            // H*W
#define N_PTS    65536           // B*H*W
#define ZQ_ELEMS 4194304
#define TILE_PTS 128
#define TILES_PER_CB 512         // N_PTS / TILE_PTS

// ---------------------------------------------------------------------------
// Prep: split codebooks into bf16 hi/lo and precompute -0.5*||e||^2 (fp32).
// ---------------------------------------------------------------------------
__global__ void fvq_prep(const float* __restrict__ cb,
                         __bf16* __restrict__ cb_hi,
                         __bf16* __restrict__ cb_lo,
                         float* __restrict__ esq_s) {
    int e = blockIdx.x * blockDim.x + threadIdx.x;
    if (e >= NUM_CB * CB_SIZE) return;
    const float* row = cb + (size_t)e * DIM;
    float s = 0.f;
    #pragma unroll 4
    for (int d = 0; d < DIM; ++d) {
        float f = row[d];
        __bf16 h = (__bf16)f;
        __bf16 l = (__bf16)(f - (float)h);
        cb_hi[(size_t)e * DIM + d] = h;
        cb_lo[(size_t)e * DIM + d] = l;
        s += f * f;
    }
    esq_s[e] = -0.5f * s;
}

// ---------------------------------------------------------------------------
// Main: per block -> one codebook k, 128 consecutive points.
// 8 waves x 16 points; each wave scans all 1024 entries with bf16 WMMA.
// ---------------------------------------------------------------------------
__global__ __launch_bounds__(256) void fvq_main(
        const float*  __restrict__ z,
        const float*  __restrict__ cb,
        const __bf16* __restrict__ cb_hi,
        const __bf16* __restrict__ cb_lo,
        const float*  __restrict__ esq_s,
        float* __restrict__ out,
        float* __restrict__ partials) {
    __shared__ __align__(32) float zT[TILE_PTS * 65];   // transposed z tile, padded
    __shared__ __align__(32) float esq_l[CB_SIZE];      // -0.5*||e||^2
    __shared__ float wsum[8];

    const int bx   = blockIdx.x;
    const int k    = bx >> 9;            // codebook
    const int tile = bx & 511;
    const int P0   = tile * TILE_PTS;    // first point of tile
    const int bb   = P0 >> 12;           // batch index (tile never crosses batch)
    const int hw0  = P0 & 4095;
    const int tid  = threadIdx.x;

    // Stage z tile (transposed to [point][dim]) — coalesced global, no LDS conflicts.
    const float* zbase = z + (size_t)bb * (256 * HW) + (size_t)(k * DIM) * HW + hw0;
    for (int i = tid; i < DIM * TILE_PTS; i += 256) {
        int d = i >> 7, p = i & 127;
        zT[p * 65 + d] = zbase[(size_t)d * HW + p];
    }
    for (int i = tid; i < CB_SIZE; i += 256)
        esq_l[i] = esq_s[k * CB_SIZE + i];
    __syncthreads();

    const int lane = tid & 31;
    const int wave = tid >> 5;
    const int nloc = lane & 15;
    const int half = lane >> 4;
    const int p    = wave * 16 + nloc;    // point owned by this lane slot

    // Build B fragments (z, bf16 hi/lo). B layout (wave32, 16-bit, KxN=32x16):
    // lane n (<16): column n, K 0..7 & 16..23 ; lane n+16: column n, K 8..15 & 24..31.
    v16bf Bhi[2], Blo[2];
    #pragma unroll
    for (int t = 0; t < 2; ++t) {
        const float* src = &zT[p * 65 + t * 32 + half * 8];
        #pragma unroll
        for (int i = 0; i < 16; ++i) {
            float f = src[i + (i >= 8 ? 8 : 0)];
            __bf16 h = (__bf16)f;
            Bhi[t][i] = h;
            Blo[t][i] = (__bf16)(f - (float)h);
        }
    }

    const __bf16* Ahi_base = cb_hi + (size_t)k * CB_SIZE * DIM;
    const __bf16* Alo_base = cb_lo + (size_t)k * CB_SIZE * DIM;
    const int o = half * 8;   // A layout: lane<16 holds K0..7 & K16..23, lane>=16 holds K8..15 & K24..31

    float best = -1e30f;
    int   bidx = 0;

    for (int it = 0; it < 64; ++it) {
        const int s_base = it * 16;
        const int row = s_base + nloc;
        const __bf16* ah = Ahi_base + (size_t)row * DIM;
        const __bf16* al = Alo_base + (size_t)row * DIM;

        v8bf ah00 = *(const v8bf*)(ah + 0  + o);
        v8bf ah01 = *(const v8bf*)(ah + 16 + o);
        v8bf ah10 = *(const v8bf*)(ah + 32 + o);
        v8bf ah11 = *(const v8bf*)(ah + 48 + o);
        v8bf al00 = *(const v8bf*)(al + 0  + o);
        v8bf al01 = *(const v8bf*)(al + 16 + o);
        v8bf al10 = *(const v8bf*)(al + 32 + o);
        v8bf al11 = *(const v8bf*)(al + 48 + o);

        v16bf Ah0, Ah1, Al0, Al1;
        #pragma unroll
        for (int i = 0; i < 8; ++i) {
            Ah0[i] = ah00[i]; Ah0[8 + i] = ah01[i];
            Ah1[i] = ah10[i]; Ah1[8 + i] = ah11[i];
            Al0[i] = al00[i]; Al0[8 + i] = al01[i];
            Al1[i] = al10[i]; Al1[8 + i] = al11[i];
        }

        // C init = -0.5*||e||^2 folded into the accumulator (layout: acc[r] <-> s_base + half*8 + r)
        v8f acc = *(const v8f*)&esq_l[s_base + half * 8];

        // score = -0.5||e||^2 + z_hi.e_hi + z_lo.e_hi + z_hi.e_lo   (fp32 accum)
        acc = __builtin_amdgcn_wmma_f32_16x16x32_bf16(false, Ah0, false, Bhi[0], (short)0, acc, false, false);
        acc = __builtin_amdgcn_wmma_f32_16x16x32_bf16(false, Ah1, false, Bhi[1], (short)0, acc, false, false);
        acc = __builtin_amdgcn_wmma_f32_16x16x32_bf16(false, Ah0, false, Blo[0], (short)0, acc, false, false);
        acc = __builtin_amdgcn_wmma_f32_16x16x32_bf16(false, Ah1, false, Blo[1], (short)0, acc, false, false);
        acc = __builtin_amdgcn_wmma_f32_16x16x32_bf16(false, Al0, false, Bhi[0], (short)0, acc, false, false);
        acc = __builtin_amdgcn_wmma_f32_16x16x32_bf16(false, Al1, false, Bhi[1], (short)0, acc, false, false);

        #pragma unroll
        for (int r = 0; r < 8; ++r) {
            float sc = acc[r];
            int si = s_base + half * 8 + r;
            if (sc > best) { best = sc; bidx = si; }
        }
    }

    // Combine the two lane halves (same point, disjoint entry subsets).
    {
        float obest = __shfl_xor(best, 16, 32);
        int   oidx  = __shfl_xor(bidx, 16, 32);
        if (obest > best || (obest == best && oidx < bidx)) { best = obest; bidx = oidx; }
    }

    // Write indices, z_q (STE value), accumulate squared error.
    float psum = 0.f;
    if (lane < 16) {
        const int n = P0 + p;
        out[ZQ_ELEMS + 1 + (size_t)k * N_PTS + n] = (float)bidx;
        const float* crow = cb + ((size_t)k * CB_SIZE + bidx) * DIM;
        float* op = out + (size_t)bb * (256 * HW) + (size_t)(k * DIM) * HW + (hw0 + p);
        const float* zrow = &zT[p * 65];
        #pragma unroll 4
        for (int d = 0; d < DIM; ++d) {
            float e  = crow[d];
            float zf = zrow[d];
            float diff = e - zf;
            op[(size_t)d * HW] = zf + diff;   // zf + stopgrad(zq - zf)
            psum += diff * diff;
        }
    }
    // Deterministic reduction: wave -> LDS -> one partial per block.
    #pragma unroll
    for (int off = 16; off > 0; off >>= 1) psum += __shfl_xor(psum, off, 32);
    if (lane == 0) wsum[wave] = psum;
    __syncthreads();
    if (tid == 0) {
        float s = 0.f;
        #pragma unroll
        for (int i = 0; i < 8; ++i) s += wsum[i];
        partials[bx] = s;
    }
}

// ---------------------------------------------------------------------------
// Final loss: fixed-order sum of block partials (deterministic, no atomics).
// total = 1.25 * SSD / (K*N*d)
// ---------------------------------------------------------------------------
__global__ void fvq_loss(const float* __restrict__ partials, float* __restrict__ out) {
    if (threadIdx.x == 0 && blockIdx.x == 0) {
        double s = 0.0;
        for (int i = 0; i < NUM_CB * TILES_PER_CB; ++i) s += (double)partials[i];
        out[ZQ_ELEMS] = (float)(s * (1.25 / 16777216.0));  // 1.25 / (4*65536*64)
    }
}

extern "C" void kernel_launch(void* const* d_in, const int* in_sizes, int n_in,
                              void* d_out, int out_size, void* d_ws, size_t ws_size,
                              hipStream_t stream) {
    (void)in_sizes; (void)n_in; (void)out_size; (void)ws_size;
    const float* z  = (const float*)d_in[0];   // [16,256,64,64]
    const float* cb = (const float*)d_in[1];   // [4,1024,64]

    // Workspace layout (~1.03 MB): cb_hi | cb_lo | esq | partials
    char*   ws      = (char*)d_ws;
    __bf16* cb_hi   = (__bf16*)ws;
    __bf16* cb_lo   = cb_hi + (size_t)NUM_CB * CB_SIZE * DIM;
    float*  esq     = (float*)(cb_lo + (size_t)NUM_CB * CB_SIZE * DIM);
    float*  partials = esq + NUM_CB * CB_SIZE;
    float*  out     = (float*)d_out;

    fvq_prep<<<16, 256, 0, stream>>>(cb, cb_hi, cb_lo, esq);
    fvq_main<<<NUM_CB * TILES_PER_CB, 256, 0, stream>>>(z, cb, cb_hi, cb_lo, esq, out, partials);
    fvq_loss<<<1, 32, 0, stream>>>(partials, out);
}
